// WaveTransformerBlock_86973087744621
// MI455X (gfx1250) — matrix operand order repaired and checked by reading, not compiled, not run on hardware
//
#include <hip/hip_runtime.h>

// ---------------------------------------------------------------------------
// Types for CDNA5 WMMA
// ---------------------------------------------------------------------------
typedef __bf16 bf16;
typedef __attribute__((ext_vector_type(16))) __bf16 v16bf;
typedef __attribute__((ext_vector_type(8)))  float  v8f;

union Frag16 { unsigned int u[8]; v16bf v; };

#define BM 128
#define BN 128
#define BK 32
#define LDSP 40   // padded LDS row pitch (bf16 elems): 80B rows, 16B aligned, conflict-free

// ---------------------------------------------------------------------------
// CDNA5 async global->LDS copy (ASYNCcnt-tracked DMA, bypasses VGPRs).
// VDST = LDS byte offset (low 32 bits of generic LDS pointer per aperture rules),
// VADDR = 64-bit global address.
// ---------------------------------------------------------------------------
__device__ __forceinline__ void async_copy_b128(void* lds_dst, const void* gsrc)
{
  unsigned loff = (unsigned)(unsigned long long)lds_dst;
  asm volatile("global_load_async_to_lds_b128 %0, %1, off"
               :: "v"(loff), "v"(gsrc)
               : "memory");
}
__device__ __forceinline__ void wait_async_le4() { asm volatile("s_wait_asynccnt 4" ::: "memory"); }
__device__ __forceinline__ void wait_async_0()  { asm volatile("s_wait_asynccnt 0" ::: "memory"); }

// ---------------------------------------------------------------------------
// bf16 WMMA GEMM: C[m][n] = sum_k A[m][k] * Bt[n][k]  (+bias +residual)
// A row-major [M][lda], Bt "B-transposed" [N][ldb] (rows are K-contiguous),
// C f32 [M][ldc]. All M,N multiples of 128; K multiple of 32 (guaranteed here).
// 256 threads = 8 wave32 as 2(M)x4(N); each wave computes 64x32 via 4x2 WMMA tiles.
// Double-buffered LDS fed by async-to-LDS DMA; compute overlaps next-tile fetch.
// ---------------------------------------------------------------------------
__global__ __launch_bounds__(256) void k_gemm_bf16(
    const bf16* __restrict__ A, const bf16* __restrict__ Bt, float* __restrict__ C,
    int K, int lda, int ldb, int ldc,
    long long bsA, long long bsB, long long bsC,
    const float* __restrict__ bias_r, const float* __restrict__ bias_i, int halfN,
    const float* __restrict__ res, long long bsR)
{
  __shared__ __align__(16) bf16 As[2][BM * LDSP];
  __shared__ __align__(16) bf16 Bs[2][BN * LDSP];

  const int bz = blockIdx.z;
  A  += (long long)bz * bsA;
  Bt += (long long)bz * bsB;
  C  += (long long)bz * bsC;
  if (res) res += (long long)bz * bsR;

  const int tileM = blockIdx.y * BM;
  const int tileN = blockIdx.x * BN;
  const int tid   = threadIdx.x;
  const int lane  = tid & 31;
  const int wv    = tid >> 5;      // 0..7
  const int waveM = wv >> 2;       // 0..1  -> 64 rows each
  const int waveN = wv & 3;        // 0..3  -> 32 cols each
  const int hf    = lane >> 4;     // lane half (holds k 16*hf..16*hf+15, contiguous)
  const int l16   = lane & 15;

  // Per-thread staging chunks: rows r0 and r0+64, 16B column chunk c0.
  const int r0 = tid >> 2;
  const int c0 = (tid & 3) << 3;

  v8f acc[4][2];
  const v8f vz = {0.f,0.f,0.f,0.f,0.f,0.f,0.f,0.f};
#pragma unroll
  for (int i = 0; i < 4; ++i) { acc[i][0] = vz; acc[i][1] = vz; }

  auto issue = [&](int k0, int b) {
    async_copy_b128(&As[b][r0 * LDSP + c0],
                    &A [(size_t)(tileM + r0) * lda + k0 + c0]);
    async_copy_b128(&As[b][(r0 + 64) * LDSP + c0],
                    &A [(size_t)(tileM + r0 + 64) * lda + k0 + c0]);
    async_copy_b128(&Bs[b][r0 * LDSP + c0],
                    &Bt[(size_t)(tileN + r0) * ldb + k0 + c0]);
    async_copy_b128(&Bs[b][(r0 + 64) * LDSP + c0],
                    &Bt[(size_t)(tileN + r0 + 64) * ldb + k0 + c0]);
  };

  const int T = K / BK;
  issue(0, 0);                       // prologue: tile 0 -> buffer 0

  for (int i = 0; i < T; ++i) {
    const int b = i & 1;
    if (i + 1 < T) {                 // overlap: fetch tile i+1 into other buffer
      issue((i + 1) * BK, b ^ 1);
      wait_async_le4();              // in-order async => tile i has landed in LDS
      __builtin_prefetch(&A [(size_t)(tileM + r0) * lda + (i + 1) * BK + c0], 0, 1);
      __builtin_prefetch(&Bt[(size_t)(tileN + r0) * ldb + (i + 1) * BK + c0], 0, 1);
    } else {
      wait_async_0();
    }
    __syncthreads();                 // all waves' copies of tile i visible

    // Fragment loads: element s of lane half hf is k = 16*hf + s (same order for A and B).
    Frag16 af[4], bq[2];
#pragma unroll
    for (int mt = 0; mt < 4; ++mt) {
      int r = waveM * 64 + mt * 16 + l16;
#pragma unroll
      for (int j = 0; j < 8; ++j) {
        int kl = 2 * j + 16 * hf;
        af[mt].u[j] = *(const unsigned int*)&As[b][r * LDSP + kl];
      }
    }
#pragma unroll
    for (int nt = 0; nt < 2; ++nt) {
      int r = waveN * 32 + nt * 16 + l16;
#pragma unroll
      for (int j = 0; j < 8; ++j) {
        int kl = 2 * j + 16 * hf;
        bq[nt].u[j] = *(const unsigned int*)&Bs[b][r * LDSP + kl];
      }
    }
#pragma unroll
    for (int mt = 0; mt < 4; ++mt)
#pragma unroll
      for (int nt = 0; nt < 2; ++nt)
        acc[mt][nt] = __builtin_amdgcn_wmma_f32_16x16x32_bf16(
            false, af[mt].v, false, bq[nt].v, (short)0, acc[mt][nt], false, false);

    __syncthreads();                 // reads of buffer b done before it is refilled
  }

  // Epilogue: C/D layout -> VGPR v holds row (v + 8*hf), lane l16 holds col.
#pragma unroll
  for (int mt = 0; mt < 4; ++mt) {
#pragma unroll
    for (int nt = 0; nt < 2; ++nt) {
      int cb = tileN + waveN * 32 + nt * 16 + l16;
      float bv = 0.f;
      if (bias_r) bv = (cb < halfN) ? bias_r[cb] : bias_i[cb - halfN];
#pragma unroll
      for (int v2 = 0; v2 < 8; ++v2) {
        int rr = tileM + waveM * 64 + mt * 16 + v2 + 8 * hf;
        float xv = acc[mt][nt][v2] + bv;
        if (res) xv += res[(size_t)rr * ldc + cb];
        C[(size_t)rr * ldc + cb] = xv;
      }
    }
  }
}

// ---------------------------------------------------------------------------
// Radix-2 length-1024 FFT in LDS. 256 threads handle 2 columns (128 thr each).
// Reads f32 (strided), writes bf16 (strided), forward or inverse (+1/N scale).
// ---------------------------------------------------------------------------
__global__ __launch_bounds__(256) void k_fft1024(
    const float* __restrict__ in, bf16* __restrict__ out,
    long long in_sb, long long in_sh, long long in_sd, long long in_st, long long in_ri,
    long long out_sb, long long out_sh, long long out_sd, long long out_st, long long out_ri,
    int inverse)
{
  __shared__ float sre[2][1024];
  __shared__ float sim[2][1024];
  const int b  = blockIdx.z, hh = blockIdx.y, dg = blockIdx.x;
  const int tid = threadIdx.x;
  const int cl = tid >> 7;       // column 0/1 within block
  const int tt = tid & 127;
  const int d  = dg * 2 + cl;
  const size_t ioff = (size_t)b * in_sb  + (size_t)hh * in_sh  + (size_t)d * in_sd;
  const size_t ooff = (size_t)b * out_sb + (size_t)hh * out_sh + (size_t)d * out_sd;

  for (int i = tt; i < 1024; i += 128) {
    unsigned rev = __brev((unsigned)i) >> 22;     // 10-bit bit-reversal
    sre[cl][rev] = in[ioff + (size_t)i * in_st];
    sim[cl][rev] = in[ioff + (size_t)i * in_st + in_ri];
  }
  __syncthreads();

  const float sgn = inverse ? 1.0f : -1.0f;
  for (int len = 2; len <= 1024; len <<= 1) {
    int hl = len >> 1;
    float base = sgn * 6.283185307179586f / (float)len;
    for (int idx = tt; idx < 512; idx += 128) {
      int grp = idx / hl;
      int j   = idx - grp * hl;
      int pos = grp * len + j;
      float s, c;
      __sincosf(base * (float)j, &s, &c);
      float ur = sre[cl][pos],       ui = sim[cl][pos];
      float wr = sre[cl][pos + hl],  wi = sim[cl][pos + hl];
      float tr = c * wr - s * wi;
      float ti = c * wi + s * wr;
      sre[cl][pos]      = ur + tr;  sim[cl][pos]      = ui + ti;
      sre[cl][pos + hl] = ur - tr;  sim[cl][pos + hl] = ui - ti;
    }
    __syncthreads();
  }

  const float scale = inverse ? (1.0f / 1024.0f) : 1.0f;
  for (int i = tt; i < 1024; i += 128) {
    out[ooff + (size_t)i * out_st]          = (bf16)(sre[cl][i] * scale);
    out[ooff + (size_t)i * out_st + out_ri] = (bf16)(sim[cl][i] * scale);
  }
}

// ---------------------------------------------------------------------------
// Row softmax (with pre-scale) -> bf16, one block per row of n cols.
// ---------------------------------------------------------------------------
__global__ __launch_bounds__(256) void k_softmax_rows(
    const float* __restrict__ S, bf16* __restrict__ Aout, int n, float scale)
{
  __shared__ float red[256];
  const size_t base = (size_t)blockIdx.x * n;
  const int tid = threadIdx.x;

  float mx = -3.4e38f;
  for (int j = tid; j < n; j += 256) mx = fmaxf(mx, S[base + j] * scale);
  red[tid] = mx; __syncthreads();
  for (int s = 128; s > 0; s >>= 1) { if (tid < s) red[tid] = fmaxf(red[tid], red[tid + s]); __syncthreads(); }
  mx = red[0]; __syncthreads();

  float sum = 0.f;
  for (int j = tid; j < n; j += 256) sum += __expf(S[base + j] * scale - mx);
  red[tid] = sum; __syncthreads();
  for (int s = 128; s > 0; s >>= 1) { if (tid < s) red[tid] += red[tid + s]; __syncthreads(); }
  float inv = 1.0f / red[0];

  for (int j = tid; j < n; j += 256)
    Aout[base + j] = (bf16)(__expf(S[base + j] * scale - mx) * inv);
}

// ---------------------------------------------------------------------------
// Elementwise / packing kernels
// ---------------------------------------------------------------------------
__global__ void k_pack_x(const float* __restrict__ xr, const float* __restrict__ xi,
                         bf16* __restrict__ xp, float* __restrict__ xf,
                         long long rows, long long D)
{
  long long total = rows * 2 * D;
  for (long long idx = (long long)blockIdx.x * 256 + threadIdx.x; idx < total;
       idx += 256LL * gridDim.x) {
    long long m = idx / (2 * D), c = idx - m * 2 * D;
    float v = (c < D) ? xr[m * D + c] : xi[m * D + (c - D)];
    xp[idx] = (bf16)v;
    xf[idx] = v;
  }
}

// Packed complex weight in B-operand layout: out[n][k], n in [0,2No), k in [0,2Ki).
// y = x @ W^T convention (also valid for symmetric U apply).
__global__ void k_pack_w(const float* __restrict__ Wr, const float* __restrict__ Wi,
                         bf16* __restrict__ out, long long No, long long Ki)
{
  long long total = 4 * No * Ki;
  for (long long idx = (long long)blockIdx.x * 256 + threadIdx.x; idx < total;
       idx += 256LL * gridDim.x) {
    long long n = idx / (2 * Ki), k = idx - n * 2 * Ki;
    float v;
    if (n < No) v = (k < Ki) ?  Wr[n * Ki + k] : -Wi[n * Ki + (k - Ki)];
    else { long long n2 = n - No;
           v = (k < Ki) ?  Wi[n2 * Ki + k] :  Wr[n2 * Ki + (k - Ki)]; }
    out[idx] = (bf16)v;
  }
}

__global__ void k_f32_to_bf16(const float* __restrict__ in, bf16* __restrict__ out, long long n)
{
  for (long long i = (long long)blockIdx.x * 256 + threadIdx.x; i < n; i += 256LL * gridDim.x)
    out[i] = (bf16)in[i];
}

__global__ void k_gelu_bf16(const float* __restrict__ in, bf16* __restrict__ out, long long n)
{
  for (long long i = (long long)blockIdx.x * 256 + threadIdx.x; i < n; i += 256LL * gridDim.x) {
    float x = in[i];
    out[i] = (bf16)(0.5f * x * (1.0f + erff(x * 0.70710678118654752f)));
  }
}

__global__ void k_build_X(const float* __restrict__ H, const float* __restrict__ dt,
                          float* __restrict__ Xf, bf16* __restrict__ Xb, long long n)
{
  float d = dt[0];
  for (long long i = (long long)blockIdx.x * 256 + threadIdx.x; i < n; i += 256LL * gridDim.x) {
    float v = H[i] * d;
    Xf[i] = v;
    Xb[i] = (bf16)v;
  }
}

// U = cos(X) - i sin(X), 5-term Taylor (||X|| ~ 0.064 -> truncation < 1e-8)
__global__ void k_build_U(const float* __restrict__ Xf, const float* __restrict__ X2,
                          const float* __restrict__ X3, const float* __restrict__ X4,
                          const float* __restrict__ X5,
                          float* __restrict__ Ur, float* __restrict__ Ui, int dim)
{
  long long n = (long long)dim * dim;
  for (long long i = (long long)blockIdx.x * 256 + threadIdx.x; i < n; i += 256LL * gridDim.x) {
    int r = (int)(i / dim), c = (int)(i - (long long)r * dim);
    float I  = (r == c) ? 1.0f : 0.0f;
    Ur[i] = I - 0.5f * X2[i] + (1.0f / 24.0f) * X4[i];
    Ui[i] = -(Xf[i] - (1.0f / 6.0f) * X3[i] + (1.0f / 120.0f) * X5[i]);
  }
}

__global__ void k_unpack_out(const float* __restrict__ yU, float* __restrict__ out,
                             long long rows, long long D)
{
  long long n = rows * D;
  for (long long i = (long long)blockIdx.x * 256 + threadIdx.x; i < n; i += 256LL * gridDim.x) {
    long long m = i / D, j = i - m * D;
    out[i]     = yU[m * 2 * D + j];        // real plane
    out[n + i] = yU[m * 2 * D + D + j];    // imag plane
  }
}

// ---------------------------------------------------------------------------
// Host orchestration
// ---------------------------------------------------------------------------
extern "C" void kernel_launch(void* const* d_in, const int* in_sizes, int n_in,
                              void* d_out, int out_size, void* d_ws, size_t ws_size,
                              hipStream_t stream)
{
  (void)in_sizes; (void)n_in; (void)out_size; (void)ws_size;
  const float* xr    = (const float*)d_in[0];
  const float* xi    = (const float*)d_in[1];
  const float* Wr_q  = (const float*)d_in[2];  const float* Wi_q = (const float*)d_in[3];
  const float* br_q  = (const float*)d_in[4];  const float* bi_q = (const float*)d_in[5];
  const float* Wr_k  = (const float*)d_in[6];  const float* Wi_k = (const float*)d_in[7];
  const float* br_k  = (const float*)d_in[8];  const float* bi_k = (const float*)d_in[9];
  const float* Wr_v  = (const float*)d_in[10]; const float* Wi_v = (const float*)d_in[11];
  const float* br_v  = (const float*)d_in[12]; const float* bi_v = (const float*)d_in[13];
  const float* Wr_o  = (const float*)d_in[14]; const float* Wi_o = (const float*)d_in[15];
  const float* br_o  = (const float*)d_in[16]; const float* bi_o = (const float*)d_in[17];
  const float* Wr_f1 = (const float*)d_in[18]; const float* Wi_f1 = (const float*)d_in[19];
  const float* br_f1 = (const float*)d_in[20]; const float* bi_f1 = (const float*)d_in[21];
  const float* Wr_f2 = (const float*)d_in[22]; const float* Wi_f2 = (const float*)d_in[23];
  const float* br_f2 = (const float*)d_in[24]; const float* bi_f2 = (const float*)d_in[25];
  const float* Hm    = (const float*)d_in[26];
  const float* dtp   = (const float*)d_in[27];
  float* outp        = (float*)d_out;

  const long long M = 2048, D = 1024, D2 = 2048, HID = 4096;

  // ---- workspace carve-out (sequential, 256B aligned, with stage reuse) ----
  char*  w   = (char*)d_ws;
  size_t off = 0;
  auto alloc = [&](size_t bytes) -> void* {
    void* p = (void*)(w + off);
    off += (bytes + 255) & ~(size_t)255;
    return p;
  };
  bf16*  mpack = (bf16*) alloc(2048ULL * 2048 * 2);    //  8MB: xpack -> Xo -> x1pack -> x2pack
  float* xf32  = (float*)alloc(2048ULL * 2048 * 4);    // 16MB: x (residual 1)
  bf16*  packW = (bf16*) alloc(8192ULL * 2048 * 2);    // 32MB: reused for all weight packs
  float* Yq    = (float*)alloc(2048ULL * 2048 * 4);    // 16MB: Q proj -> Of
  float* Yk    = (float*)alloc(2048ULL * 2048 * 4);    // 16MB: K proj -> x1
  float* Yv    = (float*)alloc(2048ULL * 2048 * 4);    // 16MB: V proj -> x2
  bf16*  Qp    = (bf16*) alloc(32ULL * 1024 * 128 * 2);//  8MB
  bf16*  Kp    = (bf16*) alloc(32ULL * 1024 * 128 * 2);//  8MB
  bf16*  Vt    = (bf16*) alloc(32ULL * 1024 * 128 * 2);//  8MB
  float* S     = (float*)alloc(32ULL * 1024 * 1024 * 4);//128MB: scores -> yU
  bf16*  attn  = (bf16*) alloc(32ULL * 1024 * 1024 * 2);// 64MB
  float* h1    = (float*)alloc(2048ULL * 8192 * 4);    // 64MB: FFN hidden -> expm scratch
  bf16*  h1p   = (bf16*) alloc(2048ULL * 8192 * 2);    // 32MB

  float* Of = Yq;  float* x1 = Yk;  float* x2 = Yv;
  bf16*  Xo = mpack;  bf16* x1p = mpack;  bf16* x2p = mpack;
  char*  hb  = (char*)h1;
  float* Xf  = (float*)(hb +  0ULL * (1 << 20));
  float* X2f = (float*)(hb +  4ULL * (1 << 20));
  float* X3f = (float*)(hb +  8ULL * (1 << 20));
  float* X4f = (float*)(hb + 12ULL * (1 << 20));
  float* X5f = (float*)(hb + 16ULL * (1 << 20));
  float* Ur  = (float*)(hb + 20ULL * (1 << 20));
  float* Ui  = (float*)(hb + 24ULL * (1 << 20));
  bf16*  Xb  = (bf16*) (hb + 28ULL * (1 << 20));
  bf16*  X2b = (bf16*) (hb + 30ULL * (1 << 20));
  bf16*  X4b = (bf16*) (hb + 32ULL * (1 << 20));
  float* yU  = S;

  auto nb = [](long long n) { long long b = (n + 255) / 256; return (int)(b > 4096 ? 4096 : b); };
  auto gemm = [&](const bf16* A, const bf16* Bt, float* C, int Mm, int Nn, int Kk,
                  int lda, int ldb, int ldc, int batch,
                  long long bsA, long long bsB, long long bsC,
                  const float* brr, const float* bii, int halfN,
                  const float* res, long long bsR) {
    dim3 g(Nn / BN, Mm / BM, batch);
    k_gemm_bf16<<<g, 256, 0, stream>>>(A, Bt, C, Kk, lda, ldb, ldc,
                                       bsA, bsB, bsC, brr, bii, halfN, res, bsR);
  };

  // 1) pack input x -> [xr|xi] bf16 (GEMM A) and f32 copy (residual)
  k_pack_x<<<nb(M * D2), 256, 0, stream>>>(xr, xi, mpack, xf32, M, D);

  // 2) Q/K/V complex projections as single packed real GEMMs
  const float* Wrs[3] = {Wr_q, Wr_k, Wr_v};
  const float* Wis[3] = {Wi_q, Wi_k, Wi_v};
  const float* brs[3] = {br_q, br_k, br_v};
  const float* bis[3] = {bi_q, bi_k, bi_v};
  float* Ys[3] = {Yq, Yk, Yv};
  for (int p = 0; p < 3; ++p) {
    k_pack_w<<<nb(4 * D * D), 256, 0, stream>>>(Wrs[p], Wis[p], packW, D, D);
    gemm(mpack, packW, Ys[p], 2048, 2048, 2048, 2048, 2048, 2048, 1, 0, 0, 0,
         brs[p], bis[p], 1024, nullptr, 0);
  }

  // 3) FFT along sequence: Y[b,t][hd | 1024+hd] -> Qp/Kp [bh][t][128], Vt [bh][128][t]
  dim3 fgrid(32, 16, 2);
  k_fft1024<<<fgrid, 256, 0, stream>>>(Yq, Qp,
      1024LL * 2048, 64, 1, 2048, 1024,
      2097152LL, 131072LL, 1, 128, 64, 0);
  k_fft1024<<<fgrid, 256, 0, stream>>>(Yk, Kp,
      1024LL * 2048, 64, 1, 2048, 1024,
      2097152LL, 131072LL, 1, 128, 64, 0);
  k_fft1024<<<fgrid, 256, 0, stream>>>(Yv, Vt,
      1024LL * 2048, 64, 1, 2048, 1024,
      2097152LL, 131072LL, 1024, 1, 65536LL, 0);

  // 4) scores.real = Qp . Kp^T  (batched over 32 (b,h))
  gemm(Qp, Kp, S, 1024, 1024, 128, 128, 128, 1024, 32,
       131072LL, 131072LL, 1048576LL, nullptr, nullptr, 0, nullptr, 0);

  // 5) softmax(scale * S) -> bf16 attn
  k_softmax_rows<<<32768, 256, 0, stream>>>(S, attn, 1024, 0.125f);

  // 6) out_f = attn . Vf  (real x complex as one 128-col GEMM)
  gemm(attn, Vt, Of, 1024, 128, 1024, 1024, 1024, 128, 32,
       1048576LL, 131072LL, 131072LL, nullptr, nullptr, 0, nullptr, 0);

  // 7) IFFT -> packed bf16 [M][2D] (Xo)
  k_fft1024<<<fgrid, 256, 0, stream>>>(Of, Xo,
      2097152LL, 131072LL, 1, 128, 64,
      1024LL * 2048, 64, 1, 2048, 1024, 1);

  // 8) O projection + residual (x1 = x + attn_out)
  k_pack_w<<<nb(4 * D * D), 256, 0, stream>>>(Wr_o, Wi_o, packW, D, D);
  gemm(Xo, packW, x1, 2048, 2048, 2048, 2048, 2048, 2048, 1, 0, 0, 0,
       br_o, bi_o, 1024, xf32, 0);

  // 9) FFN: f1 -> exact GELU -> f2 + residual
  k_f32_to_bf16<<<nb(M * D2), 256, 0, stream>>>(x1, x1p, M * D2);
  k_pack_w<<<nb(4 * HID * D), 256, 0, stream>>>(Wr_f1, Wi_f1, packW, HID, D);
  gemm(x1p, packW, h1, 2048, 8192, 2048, 2048, 2048, 8192, 1, 0, 0, 0,
       br_f1, bi_f1, 4096, nullptr, 0);
  k_gelu_bf16<<<nb(M * 8192), 256, 0, stream>>>(h1, h1p, M * 8192);
  k_pack_w<<<nb(4 * D * HID), 256, 0, stream>>>(Wr_f2, Wi_f2, packW, D, HID);
  gemm(h1p, packW, x2, 2048, 2048, 8192, 8192, 8192, 2048, 1, 0, 0, 0,
       br_f2, bi_f2, 1024, x1, 0);

  // 10) U = expm(-i H dt) = cos(X) - i sin(X), X = H*dt (symmetric -> powers symmetric)
  k_build_X<<<nb(D * D), 256, 0, stream>>>(Hm, dtp, Xf, Xb, D * D);
  gemm(Xb,  Xb,  X2f, 1024, 1024, 1024, 1024, 1024, 1024, 1, 0, 0, 0, nullptr, nullptr, 0, nullptr, 0);
  k_f32_to_bf16<<<nb(D * D), 256, 0, stream>>>(X2f, X2b, D * D);
  gemm(X2b, Xb,  X3f, 1024, 1024, 1024, 1024, 1024, 1024, 1, 0, 0, 0, nullptr, nullptr, 0, nullptr, 0);
  gemm(X2b, X2b, X4f, 1024, 1024, 1024, 1024, 1024, 1024, 1, 0, 0, 0, nullptr, nullptr, 0, nullptr, 0);
  k_f32_to_bf16<<<nb(D * D), 256, 0, stream>>>(X4f, X4b, D * D);
  gemm(X4b, Xb,  X5f, 1024, 1024, 1024, 1024, 1024, 1024, 1, 0, 0, 0, nullptr, nullptr, 0, nullptr, 0);
  k_build_U<<<nb(D * D), 256, 0, stream>>>(Xf, X2f, X3f, X4f, X5f, Ur, Ui, 1024);

  // 11) out = x2 @ U as packed real GEMM (Ur/Ui symmetric -> same packer)
  k_pack_w<<<nb(4 * D * D), 256, 0, stream>>>(Ur, Ui, packW, D, D);
  k_f32_to_bf16<<<nb(M * D2), 256, 0, stream>>>(x2, x2p, M * D2);
  gemm(x2p, packW, yU, 2048, 2048, 2048, 2048, 2048, 2048, 1, 0, 0, 0,
       nullptr, nullptr, 0, nullptr, 0);

  // 12) split into [2,B,N,D] real/imag planes
  k_unpack_out<<<nb(M * D), 256, 0, stream>>>(yU, outp, M, D);
}
